// InputAttention_25142738551472
// MI455X (gfx1250) — matrix operand order repaired and checked by reading, not compiled, not run on hardware
//
#include <hip/hip_runtime.h>
#include <hip/hip_bf16.h>

typedef __bf16 bf16_t;
typedef __attribute__((ext_vector_type(16))) __bf16 v16bf;
typedef __attribute__((ext_vector_type(8)))  float  v8f;
typedef __attribute__((ext_vector_type(8)))  int    v8i;

#define DMODEL   1024
#define HEADS    16
#define DHEAD    64
#define LCTX     9          // 1 feat row + 8 context rows
#define TILE_N   8          // feat rows per workgroup
#define CN_STRIDE 1032      // bf16 elems per LDS row (2064B: 16B aligned, bank-rotating)
#define F_STRIDE  68        // f32 elems per LDS row for K/V/Q head slices
#define O_STRIDE  72        // bf16 elems per LDS row for O head slice

// ---- fragment loaders -------------------------------------------------------
// A 16x32 bf16 fragment: lane l holds row (l&15); lanes 0-15 K=0..15,
// lanes 16-31 K=16..31  -> two 16B chunks at +0/+16B (lane-half +32B).
static __device__ __forceinline__ v16bf load_afrag(const bf16_t* p0, const bf16_t* p1) {
    int4 a = *(const int4*)p0;
    int4 b = *(const int4*)p1;
    v8i t;
    t[0]=a.x; t[1]=a.y; t[2]=a.z; t[3]=a.w;
    t[4]=b.x; t[5]=b.y; t[6]=b.z; t[7]=b.w;
    return __builtin_bit_cast(v16bf, t);
}
// B 32x16 bf16 fragment, pre-tiled in global: lane reads 32 contiguous bytes.
static __device__ __forceinline__ v16bf load_bfrag(const bf16_t* p) {
    v8i t = *(const v8i*)p;
    return __builtin_bit_cast(v16bf, t);
}
static __device__ __forceinline__ unsigned pack_bf2(float a, float b) {
    unsigned short ua = __builtin_bit_cast(unsigned short, (bf16_t)a);
    unsigned short ub = __builtin_bit_cast(unsigned short, (bf16_t)b);
    return (unsigned)ua | ((unsigned)ub << 16);
}

// ---- weight pre-tiling: fp32 [1024x1024] -> bf16 B-fragment tiles -----------
// elem t: idx=t&15, lane=(t>>4)&31, kt=(t>>9)&31, ct=t>>14
// value = W[kt*32 + 16*(lane>=16) + idx][ct*16 + (lane&15)]
__global__ __launch_bounds__(256) void convert_weights(
    const float* __restrict__ Wq, const float* __restrict__ Wk,
    const float* __restrict__ Wv, const float* __restrict__ Wo,
    bf16_t* __restrict__ dst)
{
    size_t t = (size_t)blockIdx.x * 256 + threadIdx.x;   // [0, 4<<20)
    int m = (int)(t >> 20);
    size_t e = t & ((1u << 20) - 1);
    const float* W = (m == 0) ? Wq : (m == 1) ? Wk : (m == 2) ? Wv : Wo;
    int idx = (int)(e & 15);
    int ln  = (int)((e >> 4) & 31);
    int kt  = (int)((e >> 9) & 31);
    int ct  = (int)(e >> 14);
    int row = kt * 32 + ((ln & 16) ? 16 : 0) + idx;
    int col = ct * 16 + (ln & 15);
    dst[t] = (bf16_t)W[(size_t)row * DMODEL + col];
}

// ---- fused LN + QKV + attention + out-proj ---------------------------------
__global__ __launch_bounds__(256, 1) void fused_attn(
    const float* __restrict__ feat, const float* __restrict__ context,
    const float* __restrict__ lnqg, const float* __restrict__ lnqb,
    const float* __restrict__ lncg, const float* __restrict__ lncb,
    const bf16_t* __restrict__ WqT, const bf16_t* __restrict__ WkT,
    const bf16_t* __restrict__ WvT, const bf16_t* __restrict__ WoT,
    const float* __restrict__ bout, float* __restrict__ out)
{
    // rows 0-71: cn (layernormed [f; context]); rows 72-79: qn; rows 80-87 pad
    __shared__ bf16_t sCN[88 * CN_STRIDE];      // 181,632 B
    __shared__ float  sK[80 * F_STRIDE];        //  21,760 B
    __shared__ float  sV[80 * F_STRIDE];        //  21,760 B
    __shared__ float  sQ[16 * F_STRIDE];        //   4,352 B
    __shared__ bf16_t sO[16 * O_STRIDE];        //   2,304 B
    __shared__ float  sScore[TILE_N * LCTX];
    __shared__ float  sAttn[TILE_N * LCTX];

    const int tid   = threadIdx.x;
    const int lane  = tid & 31;
    const int wave  = tid >> 5;
    const int lmod  = lane & 15;
    const int lhalf = lane >> 4;
    const int n0    = blockIdx.x * TILE_N;

    // ---------------- Phase 0: LayerNorm -> bf16 LDS image ----------------
    for (int task = wave; task < 72; task += 8) {
        const float* src; int cnRow; int isFeat = 0;
        if (task < 8) {
            src = feat + (size_t)(n0 + task) * DMODEL; cnRow = task * LCTX; isFeat = 1;
        } else {
            int idx = task - 8, n = idx >> 3, k = idx & 7;
            src = context + ((size_t)(n0 + n) * 8 + k) * DMODEL;
            cnRow = n * LCTX + 1 + k;
        }
        float4 x[8]; float s = 0.f, s2 = 0.f;
        #pragma unroll
        for (int i = 0; i < 8; ++i) {
            x[i] = *(const float4*)(src + lane * 4 + i * 128);
            s  += x[i].x + x[i].y + x[i].z + x[i].w;
            s2 += x[i].x*x[i].x + x[i].y*x[i].y + x[i].z*x[i].z + x[i].w*x[i].w;
        }
        #pragma unroll
        for (int m = 16; m >= 1; m >>= 1) {
            s  += __shfl_xor(s,  m, 32);
            s2 += __shfl_xor(s2, m, 32);
        }
        float mean = s  * (1.f / 1024.f);
        float var  = s2 * (1.f / 1024.f) - mean * mean;
        float rstd = rsqrtf(var + 1e-5f);
        #pragma unroll
        for (int i = 0; i < 8; ++i) {
            int dbase = lane * 4 + i * 128;
            float y0 = (x[i].x - mean) * rstd, y1 = (x[i].y - mean) * rstd;
            float y2 = (x[i].z - mean) * rstd, y3 = (x[i].w - mean) * rstd;
            float4 g = *(const float4*)(lncg + dbase);
            float4 b = *(const float4*)(lncb + dbase);
            uint2 p;
            p.x = pack_bf2(y0*g.x + b.x, y1*g.y + b.y);
            p.y = pack_bf2(y2*g.z + b.z, y3*g.w + b.w);
            *(uint2*)&sCN[(size_t)cnRow * CN_STRIDE + dbase] = p;
            if (isFeat) {
                float4 gq = *(const float4*)(lnqg + dbase);
                float4 bq = *(const float4*)(lnqb + dbase);
                uint2 q;
                q.x = pack_bf2(y0*gq.x + bq.x, y1*gq.y + bq.y);
                q.y = pack_bf2(y2*gq.z + bq.z, y3*gq.w + bq.w);
                *(uint2*)&sCN[(size_t)(72 + task) * CN_STRIDE + dbase] = q;
            }
        }
    }
    // zero pad rows 8-15 of sO (fed to the out-GEMM A fragments)
    {
        int r = 8 + (tid >> 5), c2 = tid & 31;
        *(unsigned*)&sO[r * O_STRIDE + c2 * 2] = 0u;
    }
    __syncthreads();

    v8f accOut[8];
    v8f vzero = {};
    #pragma unroll
    for (int i = 0; i < 8; ++i) accOut[i] = vzero;

    // ---------------- Head loop ----------------
    for (int h = 0; h < HEADS; ++h) {
        // --- Step 1: Q_h / K_h / V_h = A(16|80 rows) x W[:, h*64 : h*64+64] ---
        // 11 row-tiles x 2 column-pairs = 22 jobs over 8 waves;
        // even/odd-k split -> 4 independent WMMA chains per wave.
        for (int j = wave; j < 22; j += 8) {
            int rt = j >> 1, cp = j & 1;
            int arow; const bf16_t* Wt;
            if (rt == 0)      { arow = 72;            Wt = WqT; }
            else if (rt <= 5) { arow = (rt - 1) * 16; Wt = WkT; }
            else              { arow = (rt - 6) * 16; Wt = WvT; }
            // per-lane base; all in-loop B addresses are base + literal imm
            const bf16_t* wb = Wt + ((size_t)(h * 4 + cp * 2) * 16384) + (size_t)lane * 16;
            const bf16_t* ab = sCN + (size_t)(arow + lmod) * CN_STRIDE + lhalf * 16;
            v8f a0e = vzero, a0o = vzero, a1e = vzero, a1o = vzero;
            #pragma unroll 2
            for (int kk = 0; kk < 32; kk += 2) {
                v16bf Ae  = load_afrag(ab + kk * 32,      ab + kk * 32 + 8);
                v16bf B0e = load_bfrag(wb + (size_t)kk * 512);
                v16bf B1e = load_bfrag(wb + (size_t)kk * 512 + 16384);
                a0e = __builtin_amdgcn_wmma_f32_16x16x32_bf16(false, Ae, false, B0e,
                                                              (short)0, a0e, false, false);
                a1e = __builtin_amdgcn_wmma_f32_16x16x32_bf16(false, Ae, false, B1e,
                                                              (short)0, a1e, false, false);
                v16bf Ao  = load_afrag(ab + kk * 32 + 32, ab + kk * 32 + 40);
                v16bf B0o = load_bfrag(wb + (size_t)kk * 512 + 512);
                v16bf B1o = load_bfrag(wb + (size_t)kk * 512 + 16896);
                a0o = __builtin_amdgcn_wmma_f32_16x16x32_bf16(false, Ao, false, B0o,
                                                              (short)0, a0o, false, false);
                a1o = __builtin_amdgcn_wmma_f32_16x16x32_bf16(false, Ao, false, B1o,
                                                              (short)0, a1o, false, false);
            }
            v8f acc0 = a0e + a0o;
            v8f acc1 = a1e + a1o;
            float* dst; int drow; float scale;
            if (rt == 0)      { dst = sQ; drow = 0;             scale = 0.125f; } // 1/sqrt(64)
            else if (rt <= 5) { dst = sK; drow = (rt - 1) * 16; scale = 1.f; }
            else              { dst = sV; drow = (rt - 6) * 16; scale = 1.f; }
            #pragma unroll
            for (int r8 = 0; r8 < 8; ++r8) {
                int r = drow + r8 + lhalf * 8;
                dst[r * F_STRIDE + (cp * 2 + 0) * 16 + lmod] = acc0[r8] * scale;
                dst[r * F_STRIDE + (cp * 2 + 1) * 16 + lmod] = acc1[r8] * scale;
            }
        }
        __syncthreads();

        // --- Step 2: scores[n][l] = q_n . k_{n,l}  (scale folded into Q) ---
        if (tid < TILE_N * LCTX) {
            int n = tid / LCTX, l = tid % LCTX, r = n * LCTX + l;
            float s = 0.f;
            #pragma unroll
            for (int d = 0; d < DHEAD; d += 4) {
                float4 q = *(const float4*)&sQ[n * F_STRIDE + d];
                float4 k = *(const float4*)&sK[r * F_STRIDE + d];
                s += q.x*k.x + q.y*k.y + q.z*k.z + q.w*k.w;
            }
            sScore[n * LCTX + l] = s;
        }
        __syncthreads();

        // --- softmax over l = 0..8 ---
        if (tid < TILE_N) {
            float mx = -1e30f;
            #pragma unroll
            for (int l = 0; l < LCTX; ++l) mx = fmaxf(mx, sScore[tid * LCTX + l]);
            float e[LCTX], sum = 0.f;
            #pragma unroll
            for (int l = 0; l < LCTX; ++l) { e[l] = __expf(sScore[tid * LCTX + l] - mx); sum += e[l]; }
            float inv = 1.f / sum;
            #pragma unroll
            for (int l = 0; l < LCTX; ++l) sAttn[tid * LCTX + l] = e[l] * inv;
        }
        __syncthreads();

        // --- Step 3: O_h[n][d] = sum_l attn[n][l] * V_h[n*9+l][d] (float4) ---
        if (tid < 128) {
            int n = tid >> 4, d4 = (tid & 15) * 4;
            float4 o = {0.f, 0.f, 0.f, 0.f};
            #pragma unroll
            for (int l = 0; l < LCTX; ++l) {
                float a  = sAttn[n * LCTX + l];
                float4 v = *(const float4*)&sV[(n * LCTX + l) * F_STRIDE + d4];
                o.x += a * v.x; o.y += a * v.y; o.z += a * v.z; o.w += a * v.w;
            }
            uint2 p;
            p.x = pack_bf2(o.x, o.y);
            p.y = pack_bf2(o.z, o.w);
            *(uint2*)&sO[n * O_STRIDE + d4] = p;
        }
        __syncthreads();

        // --- Step 4: out += O_h @ Wout[h*64 : h*64+64, :] (acc in VGPRs) ---
        // Single per-head per-lane base pointer; every B load is base + literal
        // immediate (c*32768 + kk*1024 bytes), so no VALU address math sits
        // between the WMMAs (avoids WAR-hazard v_nops).
        {
            const bf16_t* wob =
                WoT + (((size_t)wave * 8 * 32 + (size_t)(h * 2)) * 32 + (size_t)lane) * 16;
            #pragma unroll
            for (int kk = 0; kk < 2; ++kk) {
                const bf16_t* ao = sO + (size_t)lmod * O_STRIDE + kk * 32 + lhalf * 16;
                v16bf A = load_afrag(ao, ao + 8);
                #pragma unroll
                for (int c = 0; c < 8; ++c) {
                    v16bf B = load_bfrag(wob + (size_t)c * 16384 + (size_t)kk * 512);
                    accOut[c] = __builtin_amdgcn_wmma_f32_16x16x32_bf16(false, A, false, B,
                                    (short)0, accOut[c], false, false);
                }
            }
        }
        // no trailing barrier needed: next head's LDS writes are fenced by the
        // step-1 barrier, and sO is rewritten only after two more barriers.
    }

    // ---------------- Epilogue: + bout + residual feat ----------------
    if (lane < 16) {
        #pragma unroll
        for (int ci = 0; ci < 8; ++ci) {
            int ct  = wave * 8 + ci;
            int col = ct * 16 + lmod;
            float bb = bout[col];
            #pragma unroll
            for (int r8 = 0; r8 < 8; ++r8) {
                int n = n0 + r8;
                out[(size_t)n * DMODEL + col] =
                    accOut[ci][r8] + bb + feat[(size_t)n * DMODEL + col];
            }
        }
    }
}

extern "C" void kernel_launch(void* const* d_in, const int* in_sizes, int n_in,
                              void* d_out, int out_size, void* d_ws, size_t ws_size,
                              hipStream_t stream) {
    (void)n_in; (void)out_size; (void)ws_size;
    const float* feat    = (const float*)d_in[0];
    const float* context = (const float*)d_in[1];
    const float* lnqg    = (const float*)d_in[2];
    const float* lnqb    = (const float*)d_in[3];
    const float* lncg    = (const float*)d_in[4];
    const float* lncb    = (const float*)d_in[5];
    const float* Wq      = (const float*)d_in[6];
    const float* Wk      = (const float*)d_in[7];
    const float* Wv      = (const float*)d_in[8];
    const float* Wout    = (const float*)d_in[9];
    const float* bout    = (const float*)d_in[10];
    float* out = (float*)d_out;
    bf16_t* wbf = (bf16_t*)d_ws;              // 4 x (1<<20) bf16 = 8 MB scratch

    const int N = in_sizes[0] / DMODEL;       // 32768

    convert_weights<<<(4u << 20) / 256, 256, 0, stream>>>(Wq, Wk, Wv, Wout, wbf);
    fused_attn<<<N / TILE_N, 256, 0, stream>>>(
        feat, context, lnqg, lnqb, lncg, lncb,
        wbf, wbf + (1u << 20), wbf + (2u << 20), wbf + (3u << 20),
        bout, out);
}
